// QAOAPredictorGAT_22522808500769
// MI455X (gfx1250) — matrix operand order repaired and checked by reading, not compiled, not run on hardware
//
#include <hip/hip_runtime.h>
#include <hip/hip_bf16.h>
#include <math.h>

typedef __attribute__((ext_vector_type(16))) _Float16 v16h;
typedef __attribute__((ext_vector_type(8)))  float    v8f;

// ---------------------------------------------------------------- utilities

__device__ __forceinline__ void atomicMaxF(float* addr, float val) {
  // Sign-aware float max via integer atomics (works with -inf init).
  if (val >= 0.f) atomicMax((int*)addr, __float_as_int(val));
  else            atomicMin((unsigned int*)addr, __float_as_uint(val));
}

__global__ void fill_k(float* __restrict__ p, long long n, float v) {
  long long i = (long long)blockIdx.x * blockDim.x + threadIdx.x;
  if (i < n) p[i] = v;
}

// -------------------------------------------------------------- batch norm

__global__ void bn_stats_k(const float* __restrict__ x, int n, int c,
                           float* __restrict__ sums) {
  __shared__ float ss[64], sq[64];
  int t = threadIdx.x;
  if (t < c) { ss[t] = 0.f; sq[t] = 0.f; }
  __syncthreads();
  long long total = (long long)n * c;
  for (long long i = (long long)blockIdx.x * blockDim.x + t; i < total;
       i += (long long)gridDim.x * blockDim.x) {
    int ch = (int)(i % c);
    float v = x[i];
    atomicAdd(&ss[ch], v);
    atomicAdd(&sq[ch], v * v);
  }
  __syncthreads();
  if (t < c) { atomicAdd(&sums[t], ss[t]); atomicAdd(&sums[c + t], sq[t]); }
}

__global__ void bn_apply_k(const float* __restrict__ x, float* __restrict__ y,
                           const float* __restrict__ residual,
                           const float* __restrict__ sums,
                           const float* __restrict__ g, const float* __restrict__ b,
                           int n, int c, int do_elu) {
  long long i = (long long)blockIdx.x * blockDim.x + threadIdx.x;
  if (i >= (long long)n * c) return;
  int ch = (int)(i % c);
  float m   = sums[ch] / (float)n;
  float var = sums[c + ch] / (float)n - m * m;
  float v = (x[i] - m) * rsqrtf(var + 1e-5f) * g[ch] + b[ch];
  if (do_elu) v = v > 0.f ? v : (expf(v) - 1.f);
  if (residual) v += residual[i];
  y[i] = v;
}

// ------------------------------------------------------------------- GEMMs

// Layer-0 feature transform: [N,3] @ [3,64]
__global__ void gemm3_k(const float* __restrict__ x3, const float* __restrict__ w,
                        float* __restrict__ y, int n) {
  long long i = (long long)blockIdx.x * blockDim.x + threadIdx.x;
  if (i >= (long long)n * 64) return;
  int nn = (int)(i >> 6), o = (int)(i & 63);
  y[i] = x3[nn * 3 + 0] * w[o] + x3[nn * 3 + 1] * w[64 + o] + x3[nn * 3 + 2] * w[128 + o];
}

// WMMA GEMM for K=64: Y[N,FOUT] = X[N,64] @ W[64,FOUT].
// Block = 128 threads (4 waves).  Block tile = 64 rows x 64 cols:
// each wave owns a 16-row strip and iterates 4 col-tiles -> 8 v_wmma/wave.
// W staged in LDS pre-swizzled into per-lane fragment order so each B operand
// is one contiguous 32B run (2x ds_load_b128, no packing movs).
template<int FOUT>
__global__ __launch_bounds__(128) void gemm_wmma64_k(
    const float* __restrict__ X, const float* __restrict__ W,
    float* __restrict__ Y, int n) {
  __shared__ _Float16 sX[64 * 72];           // padded rows: 9.0 KB
  __shared__ _Float16 sW[4 * 2 * 32 * 16];   // [ct][kb][lane][e]: 8 KB
  int t    = threadIdx.x;
  int row0 = blockIdx.x * 64;
  int col0 = blockIdx.y * 64;

  // Stage X tile (64 rows x 64 K) as f16, row stride 72 to break bank pattern
  #pragma unroll
  for (int i = 0; i < 32; i++) {
    int idx = t + i * 128;            // 0..4095
    int r = idx >> 6, c = idx & 63;
    int gr = row0 + r;
    float v = (gr < n) ? X[gr * 64 + c] : 0.f;
    sX[r * 72 + c] = (_Float16)v;
  }
  // Stage W tile (64 K x 64 cols) swizzled: sW[((ct*2+kb)*32 + lane)*16 + e]
  // holds W[(kb*32 + 16*(lane>>4) + e) * FOUT + col0 + ct*16 + (lane&15)]
  #pragma unroll
  for (int i = 0; i < 32; i++) {
    int idx  = t + i * 128;           // 0..4095
    int e    = idx & 15;
    int lane = (idx >> 4) & 31;
    int kb   = (idx >> 9) & 1;
    int ct   = idx >> 10;
    int k    = kb * 32 + 16 * (lane >> 4) + e;
    int c    = ct * 16 + (lane & 15);
    sW[idx] = (_Float16)W[k * FOUT + col0 + c];
  }
  __syncthreads();

  int wave = t >> 5, lane = t & 31;
  int lrow = lane & 15, hi = lane >> 4;
  int arow = wave * 16 + lrow;

  v8f acc[4] = {};
  #pragma unroll
  for (int kb = 0; kb < 2; kb++) {
    // A (16x32 f16): lane's elements = K in [kb*32+16*hi, +16), matching B
    v16h a;
    const _Float16* ap = &sX[arow * 72 + kb * 32 + 16 * hi];
    #pragma unroll
    for (int e = 0; e < 16; e++) a[e] = ap[e];
    #pragma unroll
    for (int ct = 0; ct < 4; ct++) {
      v16h b;
      const _Float16* bp = &sW[((ct * 2 + kb) * 32 + lane) * 16];
      #pragma unroll
      for (int e = 0; e < 16; e++) b[e] = bp[e];
      acc[ct] = __builtin_amdgcn_wmma_f32_16x16x32_f16(
          false, a, false, b, (short)0, acc[ct], false, false);
    }
  }

  // D (16x16 f32): VGPR v -> M = v + 8*hi, N = lane&15.  32-bit index math.
  if (row0 + 64 <= n) {
    #pragma unroll
    for (int ct = 0; ct < 4; ct++) {
      int col = col0 + ct * 16 + lrow;
      #pragma unroll
      for (int v = 0; v < 8; v++) {
        int gr = row0 + wave * 16 + v + 8 * hi;
        Y[gr * FOUT + col] = acc[ct][v];
      }
    }
  } else {
    #pragma unroll
    for (int ct = 0; ct < 4; ct++) {
      int col = col0 + ct * 16 + lrow;
      #pragma unroll
      for (int v = 0; v < 8; v++) {
        int gr = row0 + wave * 16 + v + 8 * hi;
        if (gr < n) Y[gr * FOUT + col] = acc[ct][v];
      }
    }
  }
}

// ------------------------------------------------------------- GAT pieces

// asrc[n,h] = sum_c xs[n,h,c]*a_s[h,c]; adst likewise
__global__ void attn_k(const float* __restrict__ xs, const float* __restrict__ a_s,
                       const float* __restrict__ a_d, float* __restrict__ asrc,
                       float* __restrict__ adst, int n, int C) {
  long long i = (long long)blockIdx.x * blockDim.x + threadIdx.x;
  if (i >= (long long)n * 8) return;
  int nn = (int)(i >> 3), h = (int)(i & 7);
  const float* xp = xs + (long long)nn * 8 * C + h * C;
  float s = 0.f, d = 0.f;
  for (int c = 0; c < C; c++) {
    float v = xp[c];
    s += v * a_s[h * C + c];
    d += v * a_d[h * C + c];
  }
  asrc[i] = s; adst[i] = d;
}

// aec[h] = sum_c We[0,h*C+c]*ae[h,c]  (edge_attr is 1-dim -> aedge = ea*aec[h])
__global__ void aec_k(const float* __restrict__ We, const float* __restrict__ ae,
                      float* __restrict__ aec, int C) {
  int h = threadIdx.x;
  if (h >= 8) return;
  float s = 0.f;
  for (int c = 0; c < C; c++) s += We[h * C + c] * ae[h * C + c];
  aec[h] = s;
}

__global__ void edge_max_k(const int* __restrict__ ei, const float* __restrict__ ea,
                           const float* __restrict__ asrc, const float* __restrict__ adst,
                           const float* __restrict__ aec, float* __restrict__ mx, int E) {
  long long i = (long long)blockIdx.x * blockDim.x + threadIdx.x;
  if (i >= (long long)E * 8) return;
  int e = (int)(i >> 3), h = (int)(i & 7);
  int s = ei[e], d = ei[E + e];
  float l = asrc[s * 8 + h] + adst[d * 8 + h] + ea[e] * aec[h];
  l = l > 0.f ? l : 0.2f * l;
  atomicMaxF(&mx[d * 8 + h], l);
}

__global__ void mx_fix_k(float* __restrict__ mx, long long n) {
  long long i = (long long)blockIdx.x * blockDim.x + threadIdx.x;
  if (i >= n) return;
  float m = mx[i];
  if (!(m >= -3.3e38f && m <= 3.3e38f)) mx[i] = 0.f;
}

__global__ void edge_exp_k(const int* __restrict__ ei, const float* __restrict__ ea,
                           const float* __restrict__ asrc, const float* __restrict__ adst,
                           const float* __restrict__ aec, const float* __restrict__ mx,
                           float* __restrict__ ez, float* __restrict__ den, int E) {
  long long i = (long long)blockIdx.x * blockDim.x + threadIdx.x;
  if (i >= (long long)E * 8) return;
  int e = (int)(i >> 3), h = (int)(i & 7);
  int s = ei[e], d = ei[E + e];
  float l = asrc[s * 8 + h] + adst[d * 8 + h] + ea[e] * aec[h];
  l = l > 0.f ? l : 0.2f * l;
  float z = expf(l - mx[d * 8 + h]);
  ez[i] = z;
  atomicAdd(&den[d * 8 + h], z);
}

// out[d,h,c] += alpha * xs[s,h,c]; one thread per (edge, head, 8-channel group)
__global__ void edge_aggr_k(const int* __restrict__ ei, const float* __restrict__ ez,
                            const float* __restrict__ den, const float* __restrict__ xs,
                            float* __restrict__ aggr, int E, int C) {
  int cgroups = C >> 3;
  long long total = (long long)E * 8 * cgroups;
  long long i = (long long)blockIdx.x * blockDim.x + threadIdx.x;
  if (i >= total) return;
  int cg = (int)(i % cgroups);
  long long eh = i / cgroups;
  int h = (int)(eh & 7), e = (int)(eh >> 3);
  int s = ei[e], d = ei[E + e];
  float alpha = ez[(long long)e * 8 + h] / fmaxf(den[d * 8 + h], 1e-16f);
  // 8 channels = two 16B-aligned float4 gathers
  const float4* xp = (const float4*)(xs + (long long)s * 8 * C + h * C + cg * 8);
  float4 v0 = xp[0], v1 = xp[1];
  float* op = aggr + (long long)d * 8 * C + h * C + cg * 8;
  atomicAdd(&op[0], alpha * v0.x); atomicAdd(&op[1], alpha * v0.y);
  atomicAdd(&op[2], alpha * v0.z); atomicAdd(&op[3], alpha * v0.w);
  atomicAdd(&op[4], alpha * v1.x); atomicAdd(&op[5], alpha * v1.y);
  atomicAdd(&op[6], alpha * v1.z); atomicAdd(&op[7], alpha * v1.w);
}

__global__ void addbias64_k(const float* __restrict__ aggr, const float* __restrict__ bias,
                            float* __restrict__ y, int n) {
  long long i = (long long)blockIdx.x * blockDim.x + threadIdx.x;
  if (i >= (long long)n * 64) return;
  y[i] = aggr[i] + bias[i & 63];
}

// layer-2 head mean: y[n,c] = mean_h aggr[n,h*64+c] + b[c]
__global__ void headmean_k(const float* __restrict__ aggr, const float* __restrict__ bias,
                           float* __restrict__ y, int n) {
  long long i = (long long)blockIdx.x * blockDim.x + threadIdx.x;
  if (i >= (long long)n * 64) return;
  int nn = (int)(i >> 6), c = (int)(i & 63);
  float s = 0.f;
  #pragma unroll
  for (int h = 0; h < 8; h++) s += aggr[(long long)nn * 512 + h * 64 + c];
  y[i] = s * 0.125f + bias[c];
}

// ---------------------------------------------------------------- pooling

__global__ void pool_k(const float* __restrict__ hfeat, const int* __restrict__ batch,
                       float* __restrict__ gsum, float* __restrict__ gmax,
                       float* __restrict__ cnt, int n) {
  long long i = (long long)blockIdx.x * blockDim.x + threadIdx.x;
  if (i >= (long long)n * 64) return;
  int nn = (int)(i >> 6), c = (int)(i & 63);
  int b = batch[nn];
  float v = hfeat[i];
  atomicAdd(&gsum[b * 64 + c], v);
  atomicMaxF(&gmax[b * 64 + c], v);
  if (c == 0) atomicAdd(&cnt[b], 1.0f);
}

// one block per graph, 128 threads
__global__ void mlp_head_k(const float* __restrict__ gsum, const float* __restrict__ gmax,
                           const float* __restrict__ cnt,
                           const float* __restrict__ fc1w, const float* __restrict__ fc1b,
                           const float* __restrict__ fc2w, const float* __restrict__ fc2b,
                           const float* __restrict__ fgw, const float* __restrict__ fgb,
                           const float* __restrict__ fbw, const float* __restrict__ fbb,
                           float* __restrict__ out) {
  int g = blockIdx.x, t = threadIdx.x;
  __shared__ float feat[128], h1[128], h2[64];
  if (t < 64) feat[t] = gsum[g * 64 + t] / fmaxf(cnt[g], 1.f);
  else        feat[t] = gmax[g * 64 + (t - 64)];
  __syncthreads();
  float a = fc1b[t];
  for (int i = 0; i < 128; i++) a += feat[i] * fc1w[i * 128 + t];
  h1[t] = a > 0.f ? a : (expf(a) - 1.f);
  __syncthreads();
  if (t < 64) {
    float s = fc2b[t];
    for (int i = 0; i < 128; i++) s += h1[i] * fc2w[i * 64 + t];
    h2[t] = s > 0.f ? s : (expf(s) - 1.f);
  }
  __syncthreads();
  if (t == 0) {
    float ga = fgb[0], be = fbb[0];
    for (int i = 0; i < 64; i++) { ga += h2[i] * fgw[i]; be += h2[i] * fbw[i]; }
    out[g * 2 + 0] = ga;
    out[g * 2 + 1] = be;
  }
}

// ------------------------------------------------------------------ driver

static inline int blocks_for(long long n, int bs) { return (int)((n + bs - 1) / bs); }

extern "C" void kernel_launch(void* const* d_in, const int* in_sizes, int n_in,
                              void* d_out, int out_size, void* d_ws, size_t ws_size,
                              hipStream_t stream) {
  const float* x      = (const float*)d_in[0];
  const int*   ei     = (const int*)d_in[1];
  const float* ea     = (const float*)d_in[2];
  const int*   batch  = (const int*)d_in[3];
  const float* inbn_g = (const float*)d_in[4];
  const float* inbn_b = (const float*)d_in[5];
  const float* W[3]   = {(const float*)d_in[6],  (const float*)d_in[14], (const float*)d_in[22]};
  const float* We[3]  = {(const float*)d_in[7],  (const float*)d_in[15], (const float*)d_in[23]};
  const float* As[3]  = {(const float*)d_in[8],  (const float*)d_in[16], (const float*)d_in[24]};
  const float* Ad[3]  = {(const float*)d_in[9],  (const float*)d_in[17], (const float*)d_in[25]};
  const float* Ae[3]  = {(const float*)d_in[10], (const float*)d_in[18], (const float*)d_in[26]};
  const float* Bi[3]  = {(const float*)d_in[11], (const float*)d_in[19], (const float*)d_in[27]};
  const float* Bg[3]  = {(const float*)d_in[12], (const float*)d_in[20], (const float*)d_in[28]};
  const float* Bb[3]  = {(const float*)d_in[13], (const float*)d_in[21], (const float*)d_in[29]};
  const float* fc1w = (const float*)d_in[30]; const float* fc1b = (const float*)d_in[31];
  const float* fc2w = (const float*)d_in[32]; const float* fc2b = (const float*)d_in[33];
  const float* fgw  = (const float*)d_in[34]; const float* fgb  = (const float*)d_in[35];
  const float* fbw  = (const float*)d_in[36]; const float* fbb  = (const float*)d_in[37];

  const int N = in_sizes[0] / 3;
  const int E = in_sizes[2];
  const int G = out_size / 2;
  float* out = (float*)d_out;

  // workspace carve-up (floats)
  float* ws = (float*)d_ws;
  size_t o = 0;
  float* xbn3 = ws + o; o += (size_t)N * 3;
  float* h    = ws + o; o += (size_t)N * 64;
  float* hn   = ws + o; o += (size_t)N * 64;
  float* xs   = ws + o; o += (size_t)N * 512;
  float* aggr = ws + o; o += (size_t)N * 512;
  float* asrc = ws + o; o += (size_t)N * 8;
  float* adst = ws + o; o += (size_t)N * 8;
  float* mx   = ws + o; o += (size_t)N * 8;
  float* den  = ws + o; o += (size_t)N * 8;
  float* ez   = ws + o; o += (size_t)E * 8;
  float* aec  = ws + o; o += 64;
  float* stat = ws + o; o += 128;
  float* gsum = ws + o; o += (size_t)G * 64;
  float* gmax = ws + o; o += (size_t)G * 64;
  float* cnt  = ws + o; o += (size_t)G;
  (void)ws_size; (void)n_in;

  const int BS = 256;
  const float NEG_INF = -__builtin_huge_valf();

  // ---- input batch norm ([N,3])
  fill_k<<<1, 128, 0, stream>>>(stat, 128, 0.f);
  bn_stats_k<<<512, BS, 0, stream>>>(x, N, 3, stat);
  bn_apply_k<<<blocks_for((long long)N * 3, BS), BS, 0, stream>>>(
      x, xbn3, nullptr, stat, inbn_g, inbn_b, N, 3, 0);

  for (int layer = 0; layer < 3; layer++) {
    const int C    = (layer < 2) ? 8 : 64;     // channels per head
    const int Fout = 8 * C;                    // 64 or 512

    // 1) node feature transform -> xs[N, Fout]
    if (layer == 0) {
      gemm3_k<<<blocks_for((long long)N * 64, BS), BS, 0, stream>>>(xbn3, W[0], xs, N);
    } else if (layer == 1) {
      dim3 grid(blocks_for(N, 64), 1);
      gemm_wmma64_k<64><<<grid, 128, 0, stream>>>(h, W[1], xs, N);
    } else {
      dim3 grid(blocks_for(N, 64), 8);
      gemm_wmma64_k<512><<<grid, 128, 0, stream>>>(h, W[2], xs, N);
    }

    // 2) attention coefficients
    attn_k<<<blocks_for((long long)N * 8, BS), BS, 0, stream>>>(xs, As[layer], Ad[layer], asrc, adst, N, C);
    aec_k<<<1, 8, 0, stream>>>(We[layer], Ae[layer], aec, C);

    // 3) init segment buffers
    fill_k<<<blocks_for((long long)N * 8, BS), BS, 0, stream>>>(mx, (long long)N * 8, NEG_INF);
    fill_k<<<blocks_for((long long)N * 8, BS), BS, 0, stream>>>(den, (long long)N * 8, 0.f);
    fill_k<<<blocks_for((long long)N * Fout, BS), BS, 0, stream>>>(aggr, (long long)N * Fout, 0.f);

    // 4) segment softmax + aggregation over edges
    edge_max_k<<<blocks_for((long long)E * 8, BS), BS, 0, stream>>>(ei, ea, asrc, adst, aec, mx, E);
    mx_fix_k<<<blocks_for((long long)N * 8, BS), BS, 0, stream>>>(mx, (long long)N * 8);
    edge_exp_k<<<blocks_for((long long)E * 8, BS), BS, 0, stream>>>(ei, ea, asrc, adst, aec, mx, ez, den, E);
    edge_aggr_k<<<blocks_for((long long)E * 8 * (C / 8), BS), BS, 0, stream>>>(ei, ez, den, xs, aggr, E, C);

    // 5) concat / head-mean + bias -> hn[N,64]
    if (layer < 2)
      addbias64_k<<<blocks_for((long long)N * 64, BS), BS, 0, stream>>>(aggr, Bi[layer], hn, N);
    else
      headmean_k<<<blocks_for((long long)N * 64, BS), BS, 0, stream>>>(aggr, Bi[layer], hn, N);

    // 6) batch norm + ELU (+ residual for layers 1,2) -> h[N,64]
    fill_k<<<1, 128, 0, stream>>>(stat, 128, 0.f);
    bn_stats_k<<<512, BS, 0, stream>>>(hn, N, 64, stat);
    const float* resid = (layer == 0) ? nullptr : h;
    bn_apply_k<<<blocks_for((long long)N * 64, BS), BS, 0, stream>>>(
        hn, h, resid, stat, Bg[layer], Bb[layer], N, 64, 1);
  }

  // ---- dual pooling per graph
  fill_k<<<blocks_for((long long)G * 64, BS), BS, 0, stream>>>(gsum, (long long)G * 64, 0.f);
  fill_k<<<blocks_for((long long)G * 64, BS), BS, 0, stream>>>(gmax, (long long)G * 64, NEG_INF);
  fill_k<<<1, 128, 0, stream>>>(cnt, G, 0.f);
  pool_k<<<blocks_for((long long)N * 64, BS), BS, 0, stream>>>(h, batch, gsum, gmax, cnt, N);

  // ---- MLP head -> out[G,2]
  mlp_head_k<<<G, 128, 0, stream>>>(gsum, gmax, cnt, fc1w, fc1b, fc2w, fc2b,
                                    fgw, fgb, fbw, fbb, out);
}
